// DenseFixedMoE_33526514713023
// MI455X (gfx1250) — matrix operand order repaired and checked by reading, not compiled, not run on hardware
//
#include <hip/hip_runtime.h>

// Problem constants (from reference setup_inputs)
#define B_ROWS 32768
#define F_DIM  512
#define H_DIM  256
#define C_DIM  100
#define C_PAD  112          // C padded to multiple of 16 for WMMA N tiles
#define E_NUM  8
#define MT     32           // rows per block
#define THREADS 256         // 8 waves (wave32)

typedef __attribute__((ext_vector_type(16))) _Float16     v16h;
typedef __attribute__((ext_vector_type(8)))  float        v8f;
typedef __attribute__((ext_vector_type(4)))  unsigned int v4u;

union HFrag { v16h v; v4u u[2]; };

__device__ __forceinline__ v8f wmma_f16(v16h a, v16h b, v8f c) {
    // D = A(16x32 f16) * B(32x16 f16) + C(16x16 f32)
    return __builtin_amdgcn_wmma_f32_16x16x32_f16(
        /*neg_a=*/false, a, /*neg_b=*/false, b,
        /*c_mod=*/(short)0, c, /*reuse_a=*/false, /*reuse_b=*/false);
}

// ---------------------------------------------------------------------------
// Prep: W1 [E][F][H] f32 -> W1h [E][H][F] f16 (N-major rows, K contiguous)
//       also zero part_sizes (runs before gate kernel in stream order)
// ---------------------------------------------------------------------------
__global__ void prep_w1_kernel(const float* __restrict__ W1,
                               _Float16* __restrict__ W1h,
                               float* __restrict__ part_sizes) {
    int t = blockIdx.x * blockDim.x + threadIdx.x;
    if (t < E_NUM) part_sizes[t] = 0.0f;
    const int total = E_NUM * H_DIM * F_DIM;
    if (t >= total) return;
    int e = t / (H_DIM * F_DIM);
    int r = t % (H_DIM * F_DIM);
    int n = r / F_DIM;       // H index
    int k = r % F_DIM;       // F index
    W1h[t] = (_Float16)W1[((size_t)e * F_DIM + k) * H_DIM + n];
}

// Prep: W2 [E][H][C] f32 -> W2h [E][C_PAD][H] f16, zero-padded classes
__global__ void prep_w2_kernel(const float* __restrict__ W2,
                               _Float16* __restrict__ W2h) {
    int t = blockIdx.x * blockDim.x + threadIdx.x;
    const int total = E_NUM * C_PAD * H_DIM;
    if (t >= total) return;
    int e = t / (C_PAD * H_DIM);
    int r = t % (C_PAD * H_DIM);
    int c = r / H_DIM;
    int h = r % H_DIM;
    W2h[t] = (c < C_DIM) ? (_Float16)W2[((size_t)e * H_DIM + h) * C_DIM + c]
                         : (_Float16)0.0f;
}

// ---------------------------------------------------------------------------
// Gating: logits = x @ Wg ; hard top-2 with weight 1/2 (softmax is monotone,
// so top-2 of logits == top-2 of probs). part_sizes accumulated as float
// (exact for counts <= 2^24; d_out is a float buffer).
// ---------------------------------------------------------------------------
__global__ __launch_bounds__(256) void gate_kernel(
    const float* __restrict__ x, const float* __restrict__ Wg,
    float* __restrict__ gate_w, float* __restrict__ part_sizes) {
    int b = blockIdx.x * blockDim.x + threadIdx.x;
    if (b >= B_ROWS) return;
    float acc[E_NUM];
#pragma unroll
    for (int e = 0; e < E_NUM; ++e) acc[e] = 0.0f;
    const float* xr = x + (size_t)b * F_DIM;
    for (int f = 0; f < F_DIM; ++f) {
        float xv = xr[f];
        const float4* wg = (const float4*)(Wg + f * E_NUM);
        float4 w0 = wg[0], w1 = wg[1];
        acc[0] += xv * w0.x; acc[1] += xv * w0.y;
        acc[2] += xv * w0.z; acc[3] += xv * w0.w;
        acc[4] += xv * w1.x; acc[5] += xv * w1.y;
        acc[6] += xv * w1.z; acc[7] += xv * w1.w;
    }
    int i1 = 0;
#pragma unroll
    for (int e = 1; e < E_NUM; ++e) if (acc[e] > acc[i1]) i1 = e;
    int i2 = (i1 == 0) ? 1 : 0;
#pragma unroll
    for (int e = 0; e < E_NUM; ++e)
        if (e != i1 && acc[e] > acc[i2]) i2 = e;
#pragma unroll
    for (int e = 0; e < E_NUM; ++e)
        gate_w[(size_t)b * E_NUM + e] = (e == i1 || e == i2) ? 0.5f : 0.0f;
    atomicAdd(&part_sizes[i1], 1.0f);
    atomicAdd(&part_sizes[i2], 1.0f);
}

// ---------------------------------------------------------------------------
// Fused MoE: per block = 32 rows, loop all experts.
//   L1: h = relu(x@W1[e]+b1)  (WMMA f16->f32, 32x256 tile, K=512)
//   L2: p = h@W2[e]+b2        (WMMA, 32x112 tile, K=256)
//   softmax over C, write preds, accumulate gate-weighted combined in regs.
// ---------------------------------------------------------------------------
__global__ __launch_bounds__(THREADS) void moe_kernel(
    const float* __restrict__ x,
    const float* __restrict__ b1g,
    const float* __restrict__ b2g,
    const _Float16* __restrict__ W1h,   // [E][H][F]
    const _Float16* __restrict__ W2h,   // [E][C_PAD][H]
    const float* __restrict__ gate_w,   // [B][E]
    float* __restrict__ combined,       // [B][C]
    float* __restrict__ preds)          // [E][B][C]
{
    __shared__ _Float16 xs[MT * F_DIM];   // 32 KB: x tile, f16
    __shared__ _Float16 hs[MT * H_DIM];   // 16 KB: hidden tile, f16
    __shared__ float    ps[MT * C_PAD];   // 14 KB: pre/post-softmax logits

    const int tid  = threadIdx.x;
    const int lane = tid & 31;
    const int wave = tid >> 5;
    const int half = lane >> 4;
    const int l16  = lane & 15;
    const int rbase = blockIdx.x * MT;

    // Stage x tile: f32 global -> f16 LDS (vectorized)
#pragma unroll
    for (int i = 0; i < (MT * F_DIM) / (THREADS * 4); ++i) {
        int off4 = (tid + i * THREADS) * 4;
        float4 v = *(const float4*)(x + (size_t)rbase * F_DIM + off4);
        union { _Float16 h[4]; uint2 u; } pk;
        pk.h[0] = (_Float16)v.x; pk.h[1] = (_Float16)v.y;
        pk.h[2] = (_Float16)v.z; pk.h[3] = (_Float16)v.w;
        *(uint2*)(xs + off4) = pk.u;
    }
    __syncthreads();

    // Combined accumulator lives in registers of the softmax phase:
    // thread t owns row m = t>>3, columns c = (t&7) + 8*i
    const int sj = tid & 7;
    const int sm = tid >> 3;
    float combr[13];
#pragma unroll
    for (int i = 0; i < 13; ++i) combr[i] = 0.0f;

    for (int e = 0; e < E_NUM; ++e) {
        // ---------------- Layer 1: 32 rows x 32 cols per wave ----------------
        const int nbase = wave * 32;
        v8f c00 = {}, c01 = {}, c10 = {}, c11 = {};
        const _Float16* w1e = W1h + (size_t)e * H_DIM * F_DIM;
        const _Float16* xr0 = xs + l16 * F_DIM;
        const _Float16* xr1 = xs + (16 + l16) * F_DIM;
#pragma unroll 2
        for (int kk = 0; kk < F_DIM; kk += 32) {
            const int ka0 = kk + 8 * half;            // A: two 8-elem K runs
            const int ka1 = kk + 16 + 8 * half;
            HFrag a0, a1, bf0, bf1;
            a0.u[0] = *(const v4u*)(xr0 + ka0);
            a0.u[1] = *(const v4u*)(xr0 + ka1);
            a1.u[0] = *(const v4u*)(xr1 + ka0);
            a1.u[1] = *(const v4u*)(xr1 + ka1);
            const int kb = kk + 8 * half;             // B: two 8-elem K runs
            const _Float16* wb0 = w1e + (size_t)(nbase + l16) * F_DIM + kb;
            const _Float16* wb1 = w1e + (size_t)(nbase + 16 + l16) * F_DIM + kb;
            bf0.u[0] = *(const v4u*)(wb0); bf0.u[1] = *(const v4u*)(wb0 + 16);
            bf1.u[0] = *(const v4u*)(wb1); bf1.u[1] = *(const v4u*)(wb1 + 16);
            c00 = wmma_f16(a0.v, bf0.v, c00);
            c01 = wmma_f16(a0.v, bf1.v, c01);
            c10 = wmma_f16(a1.v, bf0.v, c10);
            c11 = wmma_f16(a1.v, bf1.v, c11);
        }
        {   // bias + ReLU -> hs (f16)
            const int n0 = nbase + l16;
            const int n1 = nbase + 16 + l16;
            const float bv0 = b1g[e * H_DIM + n0];
            const float bv1 = b1g[e * H_DIM + n1];
#pragma unroll
            for (int r = 0; r < 8; ++r) {
                const int m0 = r + 8 * half;          // C/D layout rows
                hs[m0 * H_DIM + n0]        = (_Float16)fmaxf(c00[r] + bv0, 0.0f);
                hs[m0 * H_DIM + n1]        = (_Float16)fmaxf(c01[r] + bv1, 0.0f);
                hs[(16 + m0) * H_DIM + n0] = (_Float16)fmaxf(c10[r] + bv0, 0.0f);
                hs[(16 + m0) * H_DIM + n1] = (_Float16)fmaxf(c11[r] + bv1, 0.0f);
            }
        }
        __syncthreads();

        // ---------------- Layer 2: waves 0..6 own one 16-wide class tile ----
        if (wave < 7) {                               // wave-uniform branch
            const int n2 = wave * 16 + l16;
            v8f d0 = {}, d1 = {};
            const _Float16* w2e = W2h + ((size_t)e * C_PAD + n2) * H_DIM;
            const _Float16* hr0 = hs + l16 * H_DIM;
            const _Float16* hr1 = hs + (16 + l16) * H_DIM;
#pragma unroll
            for (int kk = 0; kk < H_DIM; kk += 32) {
                const int ka0 = kk + 8 * half;
                const int ka1 = kk + 16 + 8 * half;
                HFrag a0, a1, bf;
                a0.u[0] = *(const v4u*)(hr0 + ka0);
                a0.u[1] = *(const v4u*)(hr0 + ka1);
                a1.u[0] = *(const v4u*)(hr1 + ka0);
                a1.u[1] = *(const v4u*)(hr1 + ka1);
                const _Float16* wb = w2e + kk + 8 * half;
                bf.u[0] = *(const v4u*)(wb); bf.u[1] = *(const v4u*)(wb + 16);
                d0 = wmma_f16(a0.v, bf.v, d0);
                d1 = wmma_f16(a1.v, bf.v, d1);
            }
            const float bias = (n2 < C_DIM) ? b2g[e * C_DIM + n2] : 0.0f;
#pragma unroll
            for (int r = 0; r < 8; ++r) {
                const int m0 = r + 8 * half;
                ps[m0 * C_PAD + n2]        = d0[r] + bias;
                ps[(16 + m0) * C_PAD + n2] = d1[r] + bias;
            }
        }
        __syncthreads();

        // ---------------- Softmax over C + combine (8 threads per row) -----
        {
            const size_t brow = (size_t)(rbase + sm);
            float mx = -1e30f;
            for (int c = sj; c < C_DIM; c += 8)
                mx = fmaxf(mx, ps[sm * C_PAD + c]);
            mx = fmaxf(mx, __shfl_xor(mx, 1));
            mx = fmaxf(mx, __shfl_xor(mx, 2));
            mx = fmaxf(mx, __shfl_xor(mx, 4));
            float sum = 0.0f;
            for (int c = sj; c < C_DIM; c += 8) {
                float t = __expf(ps[sm * C_PAD + c] - mx);
                ps[sm * C_PAD + c] = t;
                sum += t;
            }
            sum += __shfl_xor(sum, 1);
            sum += __shfl_xor(sum, 2);
            sum += __shfl_xor(sum, 4);
            const float inv = 1.0f / sum;
            const float wgt = gate_w[brow * E_NUM + e];
            float* pout = preds + ((size_t)e * B_ROWS + brow) * C_DIM;
#pragma unroll
            for (int i = 0; i < 13; ++i) {
                int c = sj + 8 * i;
                if (c < C_DIM) {
                    float p = ps[sm * C_PAD + c] * inv;
                    pout[c] = p;
                    combr[i] += wgt * p;
                }
            }
        }
        __syncthreads();   // protect ps/hs before next expert
    }

    // Write combined from registers
    {
        float* crow = combined + (size_t)(rbase + sm) * C_DIM;
#pragma unroll
        for (int i = 0; i < 13; ++i) {
            int c = sj + 8 * i;
            if (c < C_DIM) crow[c] = combr[i];
        }
    }
}

// ---------------------------------------------------------------------------
extern "C" void kernel_launch(void* const* d_in, const int* in_sizes, int n_in,
                              void* d_out, int out_size, void* d_ws, size_t ws_size,
                              hipStream_t stream) {
    const float* x  = (const float*)d_in[0];
    const float* W1 = (const float*)d_in[1];
    const float* b1 = (const float*)d_in[2];
    const float* W2 = (const float*)d_in[3];
    const float* b2 = (const float*)d_in[4];
    const float* Wg = (const float*)d_in[5];
    // d_in[6] is k (==2), baked into the gate kernel.

    float* combined   = (float*)d_out;                              // [B][C]
    float* preds      = combined + (size_t)B_ROWS * C_DIM;          // [E][B][C]
    float* part_sizes = preds + (size_t)E_NUM * B_ROWS * C_DIM;     // [E]

    char* ws = (char*)d_ws;
    float*    gate_w = (float*)ws;                                  // B*E f32 (1 MB)
    _Float16* W1h = (_Float16*)(ws + (size_t)B_ROWS * E_NUM * 4);   // E*H*F f16 (2 MB)
    _Float16* W2h = W1h + (size_t)E_NUM * H_DIM * F_DIM;            // E*C_PAD*H f16 (448 KB)

    const int n1 = E_NUM * H_DIM * F_DIM;
    prep_w1_kernel<<<(n1 + 255) / 256, 256, 0, stream>>>(W1, W1h, part_sizes);
    const int n2 = E_NUM * C_PAD * H_DIM;
    prep_w2_kernel<<<(n2 + 255) / 256, 256, 0, stream>>>(W2, W2h);
    gate_kernel<<<B_ROWS / 256, 256, 0, stream>>>(x, Wg, gate_w, part_sizes);
    moe_kernel<<<B_ROWS / MT, THREADS, 0, stream>>>(
        x, b1, b2, W1h, W2h, gate_w, combined, preds);
}
